// EdgeWeightLearner_9174050144888
// MI455X (gfx1250) — matrix operand order repaired and checked by reading, not compiled, not run on hardware
//
#include <hip/hip_runtime.h>
#include <hip/hip_bf16.h>

typedef __attribute__((ext_vector_type(16))) _Float16 half16;
typedef __attribute__((ext_vector_type(8)))  float    float8;

#define C_FEAT 128

// ---------------------------------------------------------------------------
// Phase 1: y_l[n] = x[n] . W[0:128], y_r[n] = x[n] . W[128:256]
// One wave (32 lanes) computes a 16-node tile via 4 chained
// v_wmma_f32_16x16x32_f16 ops (K = 128 = 4 * 32).
// B fragment: column 0 = Wl chunk, column 1 = Wr chunk, columns 2..15 zero.
// ---------------------------------------------------------------------------
__global__ __launch_bounds__(256) void node_gemv_wmma(
    const float* __restrict__ x,   // [N,128]
    const float* __restrict__ W,   // [256] = [Wl(128) | Wr(128)]
    float* __restrict__ yl,        // [N]
    float* __restrict__ yr,        // [N]
    int nNodes, int numTiles)
{
    const int lane    = threadIdx.x & 31;
    const int wave    = threadIdx.x >> 5;
    const int tile    = blockIdx.x * 8 + wave;
    if (tile >= numTiles) return;          // wave-uniform: EXEC stays all-1s

    const int lid     = lane & 15;         // row (A) / column (B) within tile
    const int halfSel = lane >> 4;         // selects K sub-range per ISA layout

    int row = tile * 16 + lid;
    if (row >= nNodes) row = nNodes - 1;   // clamp (tiles are exact for N=100000)
    const float* __restrict__ xrow = x + (size_t)row * C_FEAT;

    float8 acc = {};

#pragma unroll
    for (int kc = 0; kc < C_FEAT; kc += 32) {
        // ---- A fragment: 16x32 f16, lane holds K = kc + 8*halfSel + {0..7}
        //      then K = kc + 16 + 8*halfSel + {0..7}  (ISA 16-bit A layout)
        const float* pa = xrow + kc + halfSel * 8;
        float4 f0 = *(const float4*)(pa +  0);
        float4 f1 = *(const float4*)(pa +  4);
        float4 f2 = *(const float4*)(pa + 16);
        float4 f3 = *(const float4*)(pa + 20);
        half16 a;
        a[0]  = (_Float16)f0.x; a[1]  = (_Float16)f0.y;
        a[2]  = (_Float16)f0.z; a[3]  = (_Float16)f0.w;
        a[4]  = (_Float16)f1.x; a[5]  = (_Float16)f1.y;
        a[6]  = (_Float16)f1.z; a[7]  = (_Float16)f1.w;
        a[8]  = (_Float16)f2.x; a[9]  = (_Float16)f2.y;
        a[10] = (_Float16)f2.z; a[11] = (_Float16)f2.w;
        a[12] = (_Float16)f3.x; a[13] = (_Float16)f3.y;
        a[14] = (_Float16)f3.z; a[15] = (_Float16)f3.w;

        // ---- B fragment: 32x16 f16, lane holds column N=lid,
        //      K = kc + 8*halfSel + {0..7}, then K = kc + 16 + 8*halfSel + {0..7}
        half16 b;
#pragma unroll
        for (int i = 0; i < 16; ++i) b[i] = (_Float16)0.0f;
        if (lid < 2) {
            const float* wp = W + lid * C_FEAT + kc + halfSel * 8;
            float4 w0 = *(const float4*)(wp +  0);
            float4 w1 = *(const float4*)(wp +  4);
            float4 w2 = *(const float4*)(wp + 16);
            float4 w3 = *(const float4*)(wp + 20);
            b[0]  = (_Float16)w0.x; b[1]  = (_Float16)w0.y;
            b[2]  = (_Float16)w0.z; b[3]  = (_Float16)w0.w;
            b[4]  = (_Float16)w1.x; b[5]  = (_Float16)w1.y;
            b[6]  = (_Float16)w1.z; b[7]  = (_Float16)w1.w;
            b[8]  = (_Float16)w2.x; b[9]  = (_Float16)w2.y;
            b[10] = (_Float16)w2.z; b[11] = (_Float16)w2.w;
            b[12] = (_Float16)w3.x; b[13] = (_Float16)w3.y;
            b[14] = (_Float16)w3.z; b[15] = (_Float16)w3.w;
        }

        acc = __builtin_amdgcn_wmma_f32_16x16x32_f16(
                  /*neg_a=*/false, a, /*neg_b=*/false, b,
                  /*c_mod=*/(short)0, acc,
                  /*reuse_a=*/false, /*reuse_b=*/false);
    }

    // D layout (32-bit C/D 16x16): VGPR v, lanes 0-15 -> (M=v,   N=lane),
    //                                      lanes 16-31 -> (M=v+8, N=lane-16).
    // Column 0 (y_l) lives in lanes 0 & 16; column 1 (y_r) in lanes 1 & 17.
    if (lid < 2) {
        float* __restrict__ dst = (lid == 0) ? yl : yr;
        const int mbase = tile * 16 + halfSel * 8;
#pragma unroll
        for (int i = 0; i < 8; ++i) {
            const int m = mbase + i;
            if (m < nNodes) dst[m] = acc[i];
        }
    }
}

// ---------------------------------------------------------------------------
// Phase 2: per edge e:
//   w(e)  = sigmoid(y_l[row[e]] + y_r[col[e]])
//   out[e] = w(e) * w(full_right_idx[e])   (second factor recomputed inline)
// ---------------------------------------------------------------------------
__global__ __launch_bounds__(256) void edge_weights_kernel(
    const int* __restrict__ ei,    // [2,E] flattened: row = ei[0:E], col = ei[E:2E]
    const int* __restrict__ fri,   // [E]
    const float* __restrict__ yl,
    const float* __restrict__ yr,
    float* __restrict__ out,
    int E)
{
    const int e = blockIdx.x * 256 + threadIdx.x;
    if (e >= E) return;

    const int j  = fri[e];
    const int r0 = ei[e];
    const int c0 = ei[E + e];
    const int r1 = ei[j];
    const int c1 = ei[E + j];

    const float l0 = yl[r0] + yr[c0];
    const float l1 = yl[r1] + yr[c1];

    const float w0 = 1.0f / (1.0f + __expf(-l0));
    const float w1 = 1.0f / (1.0f + __expf(-l1));

    out[e] = w0 * w1;
}

extern "C" void kernel_launch(void* const* d_in, const int* in_sizes, int n_in,
                              void* d_out, int out_size, void* d_ws, size_t ws_size,
                              hipStream_t stream)
{
    const float* x   = (const float*)d_in[0];   // [N,128] f32
    const int*   ei  = (const int*)  d_in[1];   // [2,E] int32 (JAX x64-off demotes int64)
    const int*   fri = (const int*)  d_in[2];   // [E]   int32
    const float* W   = (const float*)d_in[3];   // [1,256] f32
    float*       out = (float*)d_out;           // [E] f32

    const int N = in_sizes[0] / C_FEAT;         // 100000
    const int E = in_sizes[2];                  // 1600000

    float* yl = (float*)d_ws;                   // N floats
    float* yr = yl + N;                         // N floats  (800 KB total)

    const int numTiles = (N + 15) / 16;         // 6250 (exact)
    const int gemvBlocks = (numTiles + 7) / 8;  // 8 waves (tiles) per 256-thread block
    node_gemv_wmma<<<gemvBlocks, 256, 0, stream>>>(x, W, yl, yr, N, numTiles);

    const int edgeBlocks = (E + 255) / 256;
    edge_weights_kernel<<<edgeBlocks, 256, 0, stream>>>(ei, fri, yl, yr, out, E);
}
